// EnhancedSpatialGNN_28475633172520
// MI455X (gfx1250) — matrix operand order repaired and checked by reading, not compiled, not run on hardware
//
#include <hip/hip_runtime.h>
#include <hip/hip_bf16.h>
#include <math.h>

typedef __attribute__((ext_vector_type(8)))  float  v8f;
typedef __attribute__((ext_vector_type(16))) __bf16 v16bf;
typedef __attribute__((ext_vector_type(2)))  __bf16 v2bf;

union Frag16 { unsigned int u[8]; v16bf v; };

// pack two f32 -> two bf16 (RNE) using the hardware convert path
__device__ __forceinline__ unsigned int f2bf2(float lo, float hi) {
    union { v2bf b; unsigned int u; } cv;
    cv.b = (v2bf){ (__bf16)lo, (__bf16)hi };
    return cv.u;
}

__device__ __forceinline__ unsigned short f2bf(float f) {
    union { __bf16 b; unsigned short u; } cv;
    cv.b = (__bf16)f;
    return cv.u;
}

__device__ __forceinline__ float gelu_exact(float v) {
    return 0.5f * v * (1.0f + erff(v * 0.70710678118654752f));
}

// ---------------------------------------------------------------------------
// GEMM: C[M x N] = A[M x K] @ W[K x N] (+ bias), bf16 WMMA, f32 accumulate.
// Block = 256 threads (8 waves), tile = 128x64, K-step = 32.
// Wave w: col-tile ct = w&3, row-group rg = w>>2 -> row tiles rg*4 .. rg*4+3.
// 4 WMMAs / wave / K-step, B fragment reused 4x.
// Requires N % 64 == 0, K % 32 == 0 (true for all GEMMs here); M guarded.
// ---------------------------------------------------------------------------
#define LSTR 34   // ushort row stride in LDS (32 + 2 pad)
#define BM 128
#define BN 64

__global__ __launch_bounds__(256)
void k_gemm(const float* __restrict__ A, const float* __restrict__ W,
            const float* __restrict__ bias, float* __restrict__ C,
            int M, int K, int N) {
    __shared__ __align__(16) unsigned short As[BM * LSTR]; // [row][k]
    __shared__ __align__(16) unsigned short Bs[BN * LSTR]; // [n][k] (transposed)

    const int tid  = threadIdx.x;
    const int lane = tid & 31;
    const int wv   = tid >> 5;
    const int ct   = wv & 3;
    const int rg   = wv >> 2;            // 0 or 1 -> rows rg*64 .. rg*64+63
    const int m    = lane & 15;
    const int hw   = lane >> 4;
    const int m0   = blockIdx.y * BM;
    const int n0   = blockIdx.x * BN;
    const bool fullM = (m0 + BM) <= M;

    v8f acc[4];
    #pragma unroll
    for (int t = 0; t < 4; ++t) acc[t] = (v8f){0.f,0.f,0.f,0.f,0.f,0.f,0.f,0.f};

    for (int k0 = 0; k0 < K; k0 += 32) {
        // stage A tile 128x32 (f32 -> bf16), float2 loads, uint LDS writes
        if (fullM) {
            #pragma unroll 4
            for (int i = tid; i < BM * 16; i += 256) {
                int r = i >> 4, kk = (i & 15) << 1;
                float2 v2 = *(const float2*)(A + (size_t)(m0 + r) * K + (k0 + kk));
                *(unsigned int*)(&As[r * LSTR + kk]) = f2bf2(v2.x, v2.y);
            }
        } else {
            #pragma unroll 4
            for (int i = tid; i < BM * 16; i += 256) {
                int r = i >> 4, kk = (i & 15) << 1;
                unsigned int pk = 0u;
                if (m0 + r < M) {
                    float2 v2 = *(const float2*)(A + (size_t)(m0 + r) * K + (k0 + kk));
                    pk = f2bf2(v2.x, v2.y);
                }
                *(unsigned int*)(&As[r * LSTR + kk]) = pk;
            }
        }
        // stage B tile 32x64 transposed -> Bs[n][k], float2 loads along N
        #pragma unroll 4
        for (int i = tid; i < 32 * 32; i += 256) {
            int kk = i >> 5, n = (i & 31) << 1;
            float2 v2 = *(const float2*)(W + (size_t)(k0 + kk) * N + (n0 + n));
            Bs[n * LSTR + kk]        = f2bf(v2.x);
            Bs[(n + 1) * LSTR + kk]  = f2bf(v2.y);
        }
        __syncthreads();

        // B fragment: lane holds col n=m, K = hw*16 .. hw*16+15 (contiguous)
        Frag16 bf_;
        {
            const unsigned short* bp = &Bs[(ct * 16 + m) * LSTR + hw * 16];
            #pragma unroll
            for (int e = 0; e < 8; ++e)
                bf_.u[e] = *(const unsigned int*)(bp + 2 * e);
        }
        // A fragments (4 row tiles); lane holds row m of each tile
        #pragma unroll
        for (int t = 0; t < 4; ++t) {
            Frag16 af;
            const unsigned short* ap = &As[((rg * 4 + t) * 16 + m) * LSTR];
            #pragma unroll
            for (int e = 0; e < 8; ++e) {
                int kb = hw * 16 + (e << 1);
                af.u[e] = *(const unsigned int*)(ap + kb);
            }
            acc[t] = __builtin_amdgcn_wmma_f32_16x16x32_bf16(
                         false, af.v, false, bf_.v, (short)0, acc[t], false, false);
        }
        __syncthreads();
    }

    // epilogue: pointer-walking stores, guard only the single partial row-block
    const int col = n0 + ct * 16 + m;
    const float bv = bias ? bias[col] : 0.f;
    const int rbase = m0 + rg * 64 + hw * 8;
    const size_t stride = (size_t)N;
    float* cp = C + (size_t)rbase * stride + col;
    if (fullM) {
        #pragma unroll
        for (int t = 0; t < 4; ++t) {
            float* p = cp + (size_t)(t * 16) * stride;
            #pragma unroll
            for (int d = 0; d < 8; ++d) { *p = acc[t][d] + bv; p += stride; }
        }
    } else {
        #pragma unroll
        for (int t = 0; t < 4; ++t) {
            float* p = cp + (size_t)(t * 16) * stride;
            #pragma unroll
            for (int d = 0; d < 8; ++d) {
                if (rbase + t * 16 + d < M) *p = acc[t][d] + bv;
                p += stride;
            }
        }
    }
}

// ---------------------------------------------------------------------------
// Elementwise / graph kernels
// ---------------------------------------------------------------------------
__global__ void k_fill(float* p, int n, float v) {
    for (int i = blockIdx.x * blockDim.x + threadIdx.x; i < n;
         i += gridDim.x * blockDim.x) p[i] = v;
}

__global__ void k_deg_scatter(const int* __restrict__ dst, float* deg, int E) {
    int i = blockIdx.x * blockDim.x + threadIdx.x;
    if (i < E) atomicAdd(&deg[dst[i]], 1.0f);
}

__global__ void k_deg_fin(const float* __restrict__ deg, float* dis, float* dinv, int n) {
    int i = blockIdx.x * blockDim.x + threadIdx.x;
    if (i < n) { float d = deg[i]; dis[i] = rsqrtf(d); dinv[i] = 1.0f / d; }
}

// wave-per-edge: lanes cover columns -> coalesced row gather + atomic scatter
__global__ __launch_bounds__(256)
void k_edge_scatter(const float* __restrict__ xw, float* __restrict__ agg,
                    const int* __restrict__ src, const int* __restrict__ dst,
                    const float* __restrict__ dis, int E, int D) {
    int e = blockIdx.x * 8 + (threadIdx.x >> 5);
    int lane = threadIdx.x & 31;
    if (e >= E) return;
    int s = src[e], d = dst[e];
    float nrm = dis[s] * dis[d];
    const float* xr = xw + (size_t)s * D;
    float* ar = agg + (size_t)d * D;
    for (int c = lane; c < D; c += 32)
        atomicAdd(&ar[c], xr[c] * nrm);
}

// agg += xw * dinv[row] + bias[col]; D = 1 << ld (power of two)
__global__ void k_selfloop_bias(const float* __restrict__ xw, const float* __restrict__ dinv,
                                const float* __restrict__ bias, float* __restrict__ agg,
                                int Mn, int ld) {
    int total = Mn << ld;
    int dm = (1 << ld) - 1;
    for (int i = blockIdx.x * blockDim.x + threadIdx.x; i < total;
         i += gridDim.x * blockDim.x) {
        agg[i] += xw[i] * dinv[i >> ld] + bias[i & dm];
    }
}

// wave-per-row LayerNorm (+ optional exact GELU), wave32 shuffles
__global__ __launch_bounds__(256)
void k_ln(float* __restrict__ x, const float* __restrict__ g,
          const float* __restrict__ b, int Mn, int D, int dogelu) {
    int row = blockIdx.x * 8 + (threadIdx.x >> 5);
    int lane = threadIdx.x & 31;
    if (row >= Mn) return;
    float* r = x + (size_t)row * D;
    float s = 0.f, s2 = 0.f;
    for (int c = lane; c < D; c += 32) { float v = r[c]; s += v; s2 += v * v; }
    for (int o = 16; o > 0; o >>= 1) {
        s  += __shfl_xor(s,  o, 32);
        s2 += __shfl_xor(s2, o, 32);
    }
    float mean = s / D;
    float var  = s2 / D - mean * mean; if (var < 0.f) var = 0.f;
    float rs = rsqrtf(var + 1e-5f);
    for (int c = lane; c < D; c += 32) {
        float v = (r[c] - mean) * rs * g[c] + b[c];
        if (dogelu) v = gelu_exact(v);
        r[c] = v;
    }
}

__global__ void k_add_gelu(const float* __restrict__ a, const float* __restrict__ b,
                           float* __restrict__ out, int n) {
    for (int i = blockIdx.x * blockDim.x + threadIdx.x; i < n;
         i += gridDim.x * blockDim.x)
        out[i] = gelu_exact(a[i] + b[i]);
}

// column-wise mean/max/sum pooling: g[0:D]=mean, g[D:2D]=max, g[2D:3D]=sum
__global__ __launch_bounds__(256)
void k_pool(const float* __restrict__ z, float* __restrict__ g, int Mn, int D) {
    __shared__ float ssum[256], smax[256];
    int col = blockIdx.x, t = threadIdx.x;
    float s = 0.f, mx = -3.402823466e38f;
    for (int r = t; r < Mn; r += 256) {
        float v = z[(size_t)r * D + col];
        s += v; mx = fmaxf(mx, v);
    }
    ssum[t] = s; smax[t] = mx; __syncthreads();
    for (int o = 128; o > 0; o >>= 1) {
        if (t < o) { ssum[t] += ssum[t + o]; smax[t] = fmaxf(smax[t], smax[t + o]); }
        __syncthreads();
    }
    if (t == 0) { g[col] = ssum[0] / (float)Mn; g[D + col] = smax[0]; g[2 * D + col] = ssum[0]; }
}

// single-block MLP head: 384 -> 128(LN,GELU) -> 64(LN,GELU) -> odim (opt sigmoid)
__global__ __launch_bounds__(128)
void k_head(const float* __restrict__ g,
            const float* w1, const float* b1, const float* g1, const float* bb1,
            const float* w2, const float* b2, const float* g2, const float* bb2,
            const float* w3, const float* b3, int odim, int dosig,
            float* __restrict__ out) {
    __shared__ float gin[384], h1[128], h2[64], red[2];
    int t = threadIdx.x;
    for (int i = t; i < 384; i += 128) gin[i] = g[i];
    __syncthreads();
    { float a = b1[t]; for (int k = 0; k < 384; ++k) a += gin[k] * w1[k * 128 + t]; h1[t] = a; }
    __syncthreads();
    if (t == 0) {
        float s = 0.f, s2 = 0.f;
        for (int k = 0; k < 128; ++k) { s += h1[k]; s2 += h1[k] * h1[k]; }
        float m = s / 128.f, v = s2 / 128.f - m * m; if (v < 0.f) v = 0.f;
        red[0] = m; red[1] = rsqrtf(v + 1e-5f);
    }
    __syncthreads();
    h1[t] = gelu_exact((h1[t] - red[0]) * red[1] * g1[t] + bb1[t]);
    __syncthreads();
    if (t < 64) { float a = b2[t]; for (int k = 0; k < 128; ++k) a += h1[k] * w2[k * 64 + t]; h2[t] = a; }
    __syncthreads();
    if (t == 0) {
        float s = 0.f, s2 = 0.f;
        for (int k = 0; k < 64; ++k) { s += h2[k]; s2 += h2[k] * h2[k]; }
        float m = s / 64.f, v = s2 / 64.f - m * m; if (v < 0.f) v = 0.f;
        red[0] = m; red[1] = rsqrtf(v + 1e-5f);
    }
    __syncthreads();
    if (t < 64) h2[t] = gelu_exact((h2[t] - red[0]) * red[1] * g2[t] + bb2[t]);
    __syncthreads();
    if (t < odim) {
        float a = b3[t];
        for (int k = 0; k < 64; ++k) a += h2[k] * w3[k * odim + t];
        if (dosig) a = 1.f / (1.f + expf(-a));
        out[t] = a;
    }
}

// ---------------------------------------------------------------------------
// Orchestration
// ---------------------------------------------------------------------------
static inline int ilog2(int d) { int l = 0; while ((1 << l) < d) ++l; return l; }

extern "C" void kernel_launch(void* const* d_in, const int* in_sizes, int n_in,
                              void* d_out, int out_size, void* d_ws, size_t ws_size,
                              hipStream_t stream) {
    (void)n_in; (void)out_size; (void)ws_size;
    const int IN_DIM = 512;
    const int M = in_sizes[0] / IN_DIM;        // 50000 nodes
    const int E = in_sizes[61] / 2;            // 800000 edges

    const float* F[62];
    for (int i = 0; i < 61; ++i) F[i] = (const float*)d_in[i];
    const int* eidx = (const int*)d_in[61];
    const int* src = eidx;
    const int* dst = eidx + E;

    // workspace: 3 big buffers (M x 512) + degree arrays + pooled vector
    size_t MD = (size_t)M * 512;
    float* H    = (float*)d_ws;
    float* XW   = H   + MD;
    float* AGG  = XW  + MD;
    float* deg  = AGG + MD;
    float* dis  = deg + M;
    float* dinv = dis + M;
    float* gbuf = dinv + M;
    float* out  = (float*)d_out;

    auto gemm = [&](const float* A, const float* W, const float* bias, float* C,
                    int Mm, int Kk, int Nn) {
        dim3 grid(Nn / BN, (Mm + BM - 1) / BM);
        k_gemm<<<grid, 256, 0, stream>>>(A, W, bias, C, Mm, Kk, Nn);
    };
    auto zero = [&](float* p, int n) {
        k_fill<<<2048, 256, 0, stream>>>(p, n, 0.0f);
    };
    auto ln = [&](float* x, const float* g, const float* b, int D, int dogelu) {
        k_ln<<<(M + 7) / 8, 256, 0, stream>>>(x, g, b, M, D, dogelu);
    };
    auto gcn = [&](const float* xw, float* agg, const float* bias, int D) {
        zero(agg, M * D);
        k_edge_scatter<<<(E + 7) / 8, 256, 0, stream>>>(xw, agg, src, dst, dis, E, D);
        k_selfloop_bias<<<4096, 256, 0, stream>>>(xw, dinv, bias, agg, M, ilog2(D));
    };
    auto addgelu = [&](const float* a, const float* b, float* o, int n) {
        k_add_gelu<<<4096, 256, 0, stream>>>(a, b, o, n);
    };

    // ---- degrees: deg = 1 + in-degree; dinv_sqrt; dinv ----
    k_fill<<<(M + 255) / 256, 256, 0, stream>>>(deg, M, 1.0f);
    k_deg_scatter<<<(E + 255) / 256, 256, 0, stream>>>(dst, deg, E);
    k_deg_fin<<<(M + 255) / 256, 256, 0, stream>>>(deg, dis, dinv, M);

    // ---- input projection: h = gelu(ln(x @ W + b)) ----
    gemm(F[0], F[1], F[2], H, M, 512, 512);
    ln(H, F[3], F[4], 512, 1);

    // ---- block 0 (512->512, identity skip) ----
    gemm(H, F[5], nullptr, XW, M, 512, 512);       // xw1
    gcn(XW, AGG, F[6], 512);                       // o1 in AGG
    ln(AGG, F[9], F[10], 512, 1);
    gemm(AGG, F[7], nullptr, XW, M, 512, 512);     // xw2
    gcn(XW, AGG, F[8], 512);                       // o2 in AGG
    ln(AGG, F[11], F[12], 512, 0);
    addgelu(AGG, H, H, M * 512);                   // h = gelu(o2 + h)

    // ---- block 1 (512->256, linear skip) ----
    gemm(H, F[13], nullptr, XW, M, 512, 256);      // xw1
    gcn(XW, AGG, F[14], 256);                      // o1 in AGG
    ln(AGG, F[17], F[18], 256, 1);
    gemm(H, F[21], F[22], XW, M, 512, 256);        // skip in XW (h consumed)
    gemm(AGG, F[15], nullptr, H, M, 256, 256);     // xw2 in H
    gcn(H, AGG, F[16], 256);                       // o2 in AGG
    ln(AGG, F[19], F[20], 256, 0);
    addgelu(AGG, XW, H, M * 256);                  // h = gelu(o2 + skip)

    // ---- block 2 (256->128, linear skip) ----
    gemm(H, F[23], nullptr, XW, M, 256, 128);
    gcn(XW, AGG, F[24], 128);
    ln(AGG, F[27], F[28], 128, 1);
    gemm(H, F[31], F[32], XW, M, 256, 128);        // skip
    gemm(AGG, F[25], nullptr, H, M, 128, 128);     // xw2
    gcn(H, AGG, F[26], 128);
    ln(AGG, F[29], F[30], 128, 0);
    addgelu(AGG, XW, H, M * 128);

    // ---- embedding projection ----
    gemm(H, F[33], F[34], XW, M, 128, 256);
    ln(XW, F[35], F[36], 256, 1);
    gemm(XW, F[37], F[38], AGG, M, 256, 128);
    ln(AGG, F[39], F[40], 128, 0);                 // z in AGG

    // ---- pooling: [mean | max | sum] -> gbuf[384] ----
    k_pool<<<128, 256, 0, stream>>>(AGG, gbuf, M, 128);

    // ---- heads ----
    k_head<<<1, 128, 0, stream>>>(gbuf, F[41], F[42], F[43], F[44],
                                  F[45], F[46], F[47], F[48],
                                  F[49], F[50], 5, 0, out);       // logits [5]
    k_head<<<1, 128, 0, stream>>>(gbuf, F[51], F[52], F[53], F[54],
                                  F[55], F[56], F[57], F[58],
                                  F[59], F[60], 1, 1, out + 5);   // sigmoid reg [1]
}
